// CausalSelfAttention_58841051955402
// MI455X (gfx1250) — compile-verified, matrix operand order checked by reading, not run on hardware
//
#include <hip/hip_runtime.h>
#include <hip/hip_bf16.h>
#include <math.h>

// ---------------------------------------------------------------------------
// CDNA5 (gfx1250) quartet causal self-attention.
// - all matmuls: v_wmma_f32_16x16x32_bf16 (wave32, f32 accumulate)
// - GEMM staging: global_load_async_to_lds_b128 + s_wait_asynccnt (ASYNCcnt),
//   double-buffered LDS, software pipelined.
// ---------------------------------------------------------------------------

typedef __attribute__((ext_vector_type(16))) __bf16          bf16x16;
typedef __attribute__((ext_vector_type(8)))  float           f32x8;
typedef __attribute__((ext_vector_type(8)))  unsigned short  u16x8;

union FragU { bf16x16 v; u16x8 h[2]; };

__device__ __forceinline__ unsigned short f2bf(float f) {
  union { float f; unsigned int u; } a; a.f = f;
  unsigned int u = a.u;
  u += 0x7fffu + ((u >> 16) & 1u);           // round-to-nearest-even
  return (unsigned short)(u >> 16);
}

// Load one 16x32 bf16 WMMA A-fragment (or B-fragment of B^T) from a row-major
// ushort(bf16) array. Per ISA 7.12.2: lane L holds row (L%16); K = 0..7 and
// 16..23 for lanes 0-15, K = 8..15 and 24..31 for lanes 16-31.
__device__ __forceinline__ bf16x16 load_frag(const unsigned short* base,
                                             int row0, int col0, int stride,
                                             int lane) {
  const unsigned short* p =
      base + (size_t)(row0 + (lane & 15)) * stride + col0 + ((lane >> 4) << 3);
  FragU f;
  f.h[0] = *(const u16x8*)p;
  f.h[1] = *(const u16x8*)(p + 16);
  return f.v;
}

// CDNA5 async copy: 16 bytes global -> LDS, tracked by ASYNCcnt.
// lds_off = wave-relative LDS byte offset (low 32 bits of generic pointer).
__device__ __forceinline__ void async_cp16(unsigned lds_off, const void* g) {
  unsigned long long ga = (unsigned long long)g;
  asm volatile("global_load_async_to_lds_b128 %0, %1, off"
               :: "v"(lds_off), "v"(ga) : "memory");
}
__device__ __forceinline__ void wait_async_le4() {
  asm volatile("s_wait_asynccnt 0x4" ::: "memory");
}
__device__ __forceinline__ void wait_async_0() {
  asm volatile("s_wait_asynccnt 0x0" ::: "memory");
}

// ---------------------------------------------------------------------------
// f32 -> bf16 conversion
// ---------------------------------------------------------------------------
__global__ void __launch_bounds__(256)
cvt_bf16(const float* __restrict__ in, unsigned short* __restrict__ out, int n) {
  for (int i = blockIdx.x * blockDim.x + threadIdx.x; i < n;
       i += gridDim.x * blockDim.x)
    out[i] = f2bf(in[i]);
}

// ---------------------------------------------------------------------------
// Generic C = A[M,K] * W[N,K]^T  (bf16 in, f32 accumulate).
// Block tile 128x128, K-step 32, 8 waves each computing 64x32 (8 WMMA/step).
// Double-buffered LDS fed by async-to-LDS copies; each thread owns exactly
// two 16B chunks of A and two of B per K-step (no staging loop, no int64 mul).
// mode 0: bf16 out [M,N] row-major
// mode 1: bf16 out transposed to [B=2, H=16, Dh=64, T=1024]  (for V)
// mode 2: f32  out [M,N] row-major (final output)
// ---------------------------------------------------------------------------
#define LDT 40   // LDS row stride (ushorts): 80B, 16B-aligned, conflict-padded
#define BUFB (128 * LDT * 2)   // bytes per buffer (10240)

__global__ void __launch_bounds__(256)
gemm_bf16(const unsigned short* __restrict__ A,
          const unsigned short* __restrict__ W,
          void* __restrict__ Out, int M, int N, int K, int mode) {
  __shared__ __align__(16) unsigned short As[2][128 * LDT];
  __shared__ __align__(16) unsigned short Bs[2][128 * LDT];

  const int t    = threadIdx.x;
  const int lane = t & 31;
  const int w    = t >> 5;                  // wave id 0..7 (wave32)
  const int bm   = blockIdx.y * 128;
  const int bn   = blockIdx.x * 128;
  const int wm   = (w >> 2) * 64;           // 2x4 wave grid
  const int wn   = (w & 3) * 32;

  // Per-thread staging assignment: rows r0 and r0+64, column chunk cc (16B).
  const int r0 = t >> 2;
  const int cc = (t & 3) * 8;
  const unsigned short* gA0 = A + (size_t)(bm + r0) * K + cc;
  const unsigned short* gA1 = gA0 + (size_t)64 * K;
  const unsigned short* gB0 = W + (size_t)(bn + r0) * K + cc;
  const unsigned short* gB1 = gB0 + (size_t)64 * K;
  const unsigned dA0 = (unsigned)(size_t)&As[0][r0 * LDT + cc];
  const unsigned dA1 = (unsigned)(size_t)&As[0][(r0 + 64) * LDT + cc];
  const unsigned dB0 = (unsigned)(size_t)&Bs[0][r0 * LDT + cc];
  const unsigned dB1 = (unsigned)(size_t)&Bs[0][(r0 + 64) * LDT + cc];

  f32x8 acc[4][2];
  for (int mi = 0; mi < 4; mi++)
    for (int ni = 0; ni < 2; ni++)
      for (int e = 0; e < 8; e++) acc[mi][ni][e] = 0.0f;

  const int steps = K >> 5;

  // Prologue: stage tile 0 into buffer 0.
  {
    async_cp16(dA0, gA0);
    async_cp16(dA1, gA1);
    async_cp16(dB0, gB0);
    async_cp16(dB1, gB1);
  }

  for (int kt = 0; kt < steps; kt++) {
    const int cur = kt & 1;
    if (kt + 1 < steps) {
      // Issue next tile into the other buffer (consumed 2 barriers ago).
      const unsigned off = cur ? 0u : (unsigned)BUFB;
      const int k0 = (kt + 1) << 5;
      async_cp16(dA0 + off, gA0 + k0);
      async_cp16(dA1 + off, gA1 + k0);
      async_cp16(dB0 + off, gB0 + k0);
      async_cp16(dB1 + off, gB1 + k0);
      wait_async_le4();   // in-order: current tile's 4 copies have landed
    } else {
      wait_async_0();
    }
    __syncthreads();

    const unsigned short* Ab = As[cur];
    const unsigned short* Bb = Bs[cur];
    bf16x16 af[4], bf[2];
    for (int mi = 0; mi < 4; mi++) af[mi] = load_frag(Ab, wm + mi * 16, 0, LDT, lane);
    for (int ni = 0; ni < 2; ni++) bf[ni] = load_frag(Bb, wn + ni * 16, 0, LDT, lane);

    for (int mi = 0; mi < 4; mi++)
      for (int ni = 0; ni < 2; ni++)
        acc[mi][ni] = __builtin_amdgcn_wmma_f32_16x16x32_bf16(
            false, af[mi], false, bf[ni], (short)0, acc[mi][ni], false, false);
    __syncthreads();
  }

  // Epilogue. C/D layout: N = lane%16, M = vgpr + 8*(lane/16).
  for (int mi = 0; mi < 4; mi++)
    for (int ni = 0; ni < 2; ni++)
      for (int vr = 0; vr < 8; vr++) {
        int gm = bm + wm + mi * 16 + vr + ((lane >> 4) << 3);
        int gn = bn + wn + ni * 16 + (lane & 15);
        float val = acc[mi][ni][vr];
        if (mode == 2) {
          ((float*)Out)[(size_t)gm * N + gn] = val;
        } else if (mode == 1) {          // V transposed: [2,16,64,1024]
          int b_ = gm >> 10, t_ = gm & 1023, h_ = gn >> 6, d_ = gn & 63;
          ((unsigned short*)Out)[(size_t)((((b_ << 4) + h_) << 6) + d_) * 1024 + t_] =
              f2bf(val);
        } else {
          ((unsigned short*)Out)[(size_t)gm * N + gn] = f2bf(val);
        }
      }
}

// ---------------------------------------------------------------------------
// Fused quartet attention. One block = (b, h, 16-query-row tile).
// Both full score rows (16x1024 f32, x2) live in LDS (128 KB of the 320 KB
// CDNA5 WGP LDS) so full-row mean/std normalization needs no HBM traffic.
// ---------------------------------------------------------------------------
__global__ void __launch_bounds__(256)
quartet_attn(const unsigned short* __restrict__ qb,
             const unsigned short* __restrict__ kb,
             const unsigned short* __restrict__ q2b,
             const unsigned short* __restrict__ k2b,
             const unsigned short* __restrict__ vT,   // [2,16,64,1024] bf16
             const float* __restrict__ mixture,
             const float* __restrict__ qscale,
             unsigned short* __restrict__ yb) {
  extern __shared__ char smem[];
  float*          S1   = (float*)smem;                        // 16*1024 f32
  float*          S2   = (float*)(smem + 65536);              // 16*1024 f32
  unsigned short* P    = (unsigned short*)(smem + 65536);     // aliases S2 (bf16 probs)
  float*          YACC = (float*)(smem + 65536 + 32768);      // aliases S2 tail, 16*64
  float*          RED  = (float*)(smem + 131072);             // 4*16*16
  float*          STAT = (float*)(smem + 131072 + 4096);      // 6*16

  const int t    = threadIdx.x;
  const int lane = t & 31;
  const int w    = t >> 5;
  const int qt   = blockIdx.x & 63;
  const int bh   = blockIdx.x >> 6;
  const int b    = bh >> 4, h = bh & 15;
  const int qbase = qt * 16;
  const int tok0  = b * 1024 + qbase;      // query row base in [2048,1024] arrays
  const int hcol  = h * 64;

  // Preload q / q2 fragments (Dh=64 -> 2 K-steps of 32).
  bf16x16 qA[2], q2A[2];
  for (int kk = 0; kk < 2; kk++) {
    qA[kk]  = load_frag(qb,  tok0, hcol + kk * 32, 1024, lane);
    q2A[kk] = load_frag(q2b, tok0, hcol + kk * 32, 1024, lane);
  }

  // ---- Phase 1: s1 = q k^T * scale, s2 = q2 k2^T * scale -> LDS -----------
  const float scale = 0.125f;              // 1/sqrt(64)
  for (int j = 0; j < 8; j++) {            // each wave owns 128 key columns
    int keyb = w * 128 + j * 16;
    f32x8 a1, a2;
    for (int e = 0; e < 8; e++) { a1[e] = 0.0f; a2[e] = 0.0f; }
    for (int kk = 0; kk < 2; kk++) {
      bf16x16 bk  = load_frag(kb,  b * 1024 + keyb, hcol + kk * 32, 1024, lane);
      a1 = __builtin_amdgcn_wmma_f32_16x16x32_bf16(false, qA[kk],  false, bk,
                                                   (short)0, a1, false, false);
      bf16x16 bk2 = load_frag(k2b, b * 1024 + keyb, hcol + kk * 32, 1024, lane);
      a2 = __builtin_amdgcn_wmma_f32_16x16x32_bf16(false, q2A[kk], false, bk2,
                                                   (short)0, a2, false, false);
    }
    int nn = keyb + (lane & 15);
    for (int vr = 0; vr < 8; vr++) {
      int mm = vr + ((lane >> 4) << 3);
      S1[mm * 1024 + nn] = a1[vr] * scale;
      S2[mm * 1024 + nn] = a2[vr] * scale;
    }
  }
  __syncthreads();

  // ---- Phase 2: full-row mean / unbiased std (over all 1024 cols) ---------
  {
    int r = t >> 4, seg = t & 15, cb = seg * 64;
    float s1 = 0, ss1 = 0, s2 = 0, ss2 = 0;
    for (int c = 0; c < 64; c++) {
      float v1 = S1[r * 1024 + cb + c]; s1 += v1; ss1 += v1 * v1;
      float v2 = S2[r * 1024 + cb + c]; s2 += v2; ss2 += v2 * v2;
    }
    RED[0 * 256 + r * 16 + seg] = s1;
    RED[1 * 256 + r * 16 + seg] = ss1;
    RED[2 * 256 + r * 16 + seg] = s2;
    RED[3 * 256 + r * 16 + seg] = ss2;
  }
  __syncthreads();
  if (t < 16) {
    float s1 = 0, ss1 = 0, s2 = 0, ss2 = 0;
    for (int seg = 0; seg < 16; seg++) {
      s1 += RED[0 * 256 + t * 16 + seg]; ss1 += RED[1 * 256 + t * 16 + seg];
      s2 += RED[2 * 256 + t * 16 + seg]; ss2 += RED[3 * 256 + t * 16 + seg];
    }
    float mu1  = s1 * (1.0f / 1024.0f);
    float var1 = fmaxf((ss1 - 1024.0f * mu1 * mu1) * (1.0f / 1023.0f), 0.0f);
    float mu2  = s2 * (1.0f / 1024.0f);
    float var2 = fmaxf((ss2 - 1024.0f * mu2 * mu2) * (1.0f / 1023.0f), 0.0f);
    STAT[0 * 16 + t] = mu1;
    STAT[1 * 16 + t] = 1.0f / (sqrtf(var1) + 1e-5f);
    STAT[2 * 16 + t] = mu2;
    STAT[3 * 16 + t] = 1.0f / (sqrtf(var2) + 1e-5f);
  }
  __syncthreads();

  // ---- Phase 3: normalize + sigmoid-gated mixture + causal mask + rowmax --
  const float m  = 1.0f / (1.0f + __expf(-mixture[0]));
  const float qs = qscale[0];
  {
    int r = t >> 4, seg = t & 15, cb = seg * 64;
    float mu1 = STAT[r], r1 = STAT[16 + r], mu2 = STAT[32 + r], r2 = STAT[48 + r];
    int qg = qbase + r;
    float mx = -INFINITY;
    for (int c = 0; c < 64; c++) {
      int col = cb + c;
      float n1 = (S1[r * 1024 + col] - mu1) * r1;
      float n2 = (S2[r * 1024 + col] - mu2) * r2;
      float sc = (1.0f - m) * n1 + m * n1 * n2 * qs;
      if (col > qg) sc = -INFINITY;
      S1[r * 1024 + col] = sc;
      mx = fmaxf(mx, sc);
    }
    RED[r * 16 + seg] = mx;
  }
  __syncthreads();
  if (t < 16) {
    float mx = -INFINITY;
    for (int seg = 0; seg < 16; seg++) mx = fmaxf(mx, RED[t * 16 + seg]);
    STAT[64 + t] = mx;
  }
  __syncthreads();

  // ---- Phase 4: exp + rowsum ---------------------------------------------
  {
    int r = t >> 4, seg = t & 15, cb = seg * 64;
    float mx = STAT[64 + r];
    float s = 0.0f;
    for (int c = 0; c < 64; c++) {
      int col = cb + c;
      float e = __expf(S1[r * 1024 + col] - mx);
      S1[r * 1024 + col] = e;
      s += e;
    }
    RED[r * 16 + seg] = s;
  }
  __syncthreads();
  if (t < 16) {
    float s = 0.0f;
    for (int seg = 0; seg < 16; seg++) s += RED[t * 16 + seg];
    STAT[80 + t] = 1.0f / s;
  }
  __syncthreads();

  // ---- Phase 5: probabilities -> bf16 in LDS (S2 is dead; reuse as P) -----
  {
    int r = t >> 4, seg = t & 15, cb = seg * 64;
    float rs = STAT[80 + r];
    for (int c = 0; c < 64; c++) {
      int col = cb + c;
      P[r * 1024 + col] = f2bf(S1[r * 1024 + col] * rs);
    }
  }
  for (int i = t; i < 1024; i += 256) YACC[i] = 0.0f;
  __syncthreads();

  // ---- Phase 6: y = P[16x1024] @ V[1024x64] via WMMA (V^T rows contiguous) -
  {
    int ntile = w & 3;                     // which 16-wide Dh slice
    int khalf = w >> 2;                    // split K across wave pairs
    const unsigned short* vbase = vT + (size_t)bh * 64 * 1024;
    f32x8 acc;
    for (int e = 0; e < 8; e++) acc[e] = 0.0f;
    for (int ks = 0; ks < 16; ks++) {
      int kt = (khalf * 16 + ks) * 32;
      bf16x16 ap = load_frag(P, 0, kt, 1024, lane);
      bf16x16 bv = load_frag(vbase, ntile * 16, kt, 1024, lane);
      acc = __builtin_amdgcn_wmma_f32_16x16x32_bf16(false, ap, false, bv,
                                                    (short)0, acc, false, false);
    }
    int nn = ntile * 16 + (lane & 15);
    for (int vr = 0; vr < 8; vr++) {
      int mm = vr + ((lane >> 4) << 3);
      atomicAdd(&YACC[mm * 64 + nn], acc[vr]);   // ds_add_f32 reduction
    }
  }
  __syncthreads();

  // ---- Phase 7: write y tile (bf16) back to [2048,1024] layout ------------
  for (int i = t; i < 1024; i += 256) {
    int r = i >> 6, d = i & 63;
    yb[(size_t)(tok0 + r) * 1024 + hcol + d] = f2bf(YACC[i]);
  }
}

// ---------------------------------------------------------------------------
// Host launcher
// ---------------------------------------------------------------------------
extern "C" void kernel_launch(void* const* d_in, const int* in_sizes, int n_in,
                              void* d_out, int out_size, void* d_ws, size_t ws_size,
                              hipStream_t stream) {
  const float* x   = (const float*)d_in[0];
  const float* Wq  = (const float*)d_in[1];
  const float* Wk  = (const float*)d_in[2];
  const float* Wv  = (const float*)d_in[3];
  const float* Wq2 = (const float*)d_in[4];
  const float* Wk2 = (const float*)d_in[5];
  const float* Wo  = (const float*)d_in[6];
  const float* mixture = (const float*)d_in[7];
  const float* qscale  = (const float*)d_in[8];

  const size_t NX = 2048 * 1024;   // x / projection elements
  const size_t NW = 1024 * 1024;   // weight elements

  unsigned short* ws   = (unsigned short*)d_ws;
  unsigned short* xb   = ws;             // 2M ushorts
  unsigned short* Wqb  = xb  + NX;
  unsigned short* Wkb  = Wqb + NW;
  unsigned short* Wvb  = Wkb + NW;
  unsigned short* Wq2b = Wvb + NW;
  unsigned short* Wk2b = Wq2b + NW;
  unsigned short* Wob  = Wk2b + NW;
  unsigned short* qbuf  = Wob  + NW;
  unsigned short* kbuf  = qbuf  + NX;
  unsigned short* q2buf = kbuf  + NX;
  unsigned short* k2buf = q2buf + NX;
  unsigned short* vTbuf = k2buf + NX;    // [2,16,64,1024]
  unsigned short* ybuf  = vTbuf + NX;

  cvt_bf16<<<1024, 256, 0, stream>>>(x,   xb,   (int)NX);
  cvt_bf16<<<512,  256, 0, stream>>>(Wq,  Wqb,  (int)NW);
  cvt_bf16<<<512,  256, 0, stream>>>(Wk,  Wkb,  (int)NW);
  cvt_bf16<<<512,  256, 0, stream>>>(Wv,  Wvb,  (int)NW);
  cvt_bf16<<<512,  256, 0, stream>>>(Wq2, Wq2b, (int)NW);
  cvt_bf16<<<512,  256, 0, stream>>>(Wk2, Wk2b, (int)NW);
  cvt_bf16<<<512,  256, 0, stream>>>(Wo,  Wob,  (int)NW);

  dim3 gg(1024 / 128, 2048 / 128);   // (N tiles, M tiles)
  gemm_bf16<<<gg, 256, 0, stream>>>(xb, Wqb,  qbuf,  2048, 1024, 1024, 0);
  gemm_bf16<<<gg, 256, 0, stream>>>(xb, Wkb,  kbuf,  2048, 1024, 1024, 0);
  gemm_bf16<<<gg, 256, 0, stream>>>(xb, Wq2b, q2buf, 2048, 1024, 1024, 0);
  gemm_bf16<<<gg, 256, 0, stream>>>(xb, Wk2b, k2buf, 2048, 1024, 1024, 0);
  gemm_bf16<<<gg, 256, 0, stream>>>(xb, Wvb,  vTbuf, 2048, 1024, 1024, 1);

  // 2048 blocks: (b,h) x 64 query tiles; 132.4 KB dynamic LDS each.
  quartet_attn<<<2048, 256, 135552, stream>>>(qbuf, kbuf, q2buf, k2buf, vTbuf,
                                              mixture, qscale, ybuf);

  gemm_bf16<<<gg, 256, 0, stream>>>(ybuf, Wob, d_out, 2048, 1024, 1024, 2);
}
